// Transformer_34961033790145
// MI455X (gfx1250) — compile-verified, matrix operand order checked by reading, not run on hardware
//
#include <hip/hip_runtime.h>
#include <math.h>

// ---------------------------------------------------------------------------
// CDNA5 / gfx1250 transformer decoder layer.
// Matrix math on v_wmma_f32_16x16x32_bf16; tile staging via
// global_load_async_to_lds_b128 (ASYNCcnt); reductions via ds_swizzle_b32.
// ---------------------------------------------------------------------------

typedef __attribute__((ext_vector_type(16))) __bf16 v16bf;
typedef __attribute__((ext_vector_type(8)))  float  v8f;

union BfFrag {
    v16bf v;
    unsigned int u[8];
};

__device__ __forceinline__ unsigned short f2bf(float f) {
    unsigned int x = __float_as_uint(f);
    unsigned int r = (x + 0x7FFFu + ((x >> 16) & 1u)) >> 16;  // RNE
    return (unsigned short)r;
}

__device__ __forceinline__ v8f wmma_bf16(v16bf a, v16bf b, v8f c) {
    return __builtin_amdgcn_wmma_f32_16x16x32_bf16(false, a, false, b,
                                                   (short)0, c, false, false);
}

// ds_swizzle XOR step within 32 lanes (masks <16 stay inside 16-lane halves).
template <int XM>
__device__ __forceinline__ float swzx(float x) {
    return __int_as_float(
        __builtin_amdgcn_ds_swizzle(__float_as_int(x), (XM << 10) | 0x1F));
}

// Async copy of 16 bytes global -> LDS (no VGPR data path, ASYNCcnt-tracked).
__device__ __forceinline__ void async_b128(const void* gp, void* lp) {
    unsigned lds = (unsigned)(size_t)lp;  // addr[31:0] == LDS byte offset
    asm volatile("global_load_async_to_lds_b128 %0, %1, off"
                 :
                 : "v"(lds), "v"(gp)
                 : "memory");
}

__device__ __forceinline__ void wait_async0() {
#if __has_builtin(__builtin_amdgcn_s_wait_asynccnt)
    __builtin_amdgcn_s_wait_asynccnt(0);
#else
    asm volatile("s_wait_asynccnt 0x0" ::: "memory");
#endif
}

// Load a 16x32 (A: MxK) or 32x16 (B: KxN, row index = N column) bf16 fragment.
// ISA layout: lane&15 selects row/col; half-wave selects K group; K packed in
// pairs per u32: vgpr0..3 hold K = (hi?8:0)+{0..7}, vgpr4..7 hold +16.
__device__ __forceinline__ v16bf load_frag16(const unsigned short* base,
                                             int row0, int stride) {
    int lane = threadIdx.x & 31;
    int rr   = row0 + (lane & 15);
    int k0   = ((lane >> 4) & 1) << 3;
    const unsigned short* p = base + (size_t)rr * stride + k0;
    BfFrag f;
#pragma unroll
    for (int i = 0; i < 4; ++i) {
        f.u[i]     = *(const unsigned int*)(p + 2 * i);
        f.u[4 + i] = *(const unsigned int*)(p + 16 + 2 * i);
    }
    return f.v;
}

// ---------------------------------------------------------------------------
// f32 -> bf16 conversion
// ---------------------------------------------------------------------------
__global__ void Transformer_34961033790145_cvt(const float* __restrict__ src,
                                               unsigned short* __restrict__ dst,
                                               int n) {
    for (int i = blockIdx.x * blockDim.x + threadIdx.x; i < n;
         i += gridDim.x * blockDim.x)
        dst[i] = f2bf(src[i]);
}

// ---------------------------------------------------------------------------
// GEMM: out[M x N] = A[M x K](bf16) @ W[N x K](bf16)^T + bias, opt ReLU.
// Block = 256 threads (8 waves, 4x2). Tile 128x128, BK=32.
// Each wave: 32x64 strip = 8 WMMAs per K-step. Tiles staged via async LDS DMA.
// ---------------------------------------------------------------------------
template <int OUT_BF16, int RELU>
__global__ __launch_bounds__(256) void Transformer_34961033790145_gemm(
    const unsigned short* __restrict__ A, const unsigned short* __restrict__ W,
    const float* __restrict__ bias, void* __restrict__ out, int M, int N, int K) {
    __shared__ unsigned short As[128 * 32];
    __shared__ unsigned short Ws[128 * 32];

    const int tid  = threadIdx.x;
    const int wave = tid >> 5;
    const int lane = tid & 31;
    const int wm   = (wave >> 1) * 32;  // 0,32,64,96
    const int wn   = (wave & 1) * 64;   // 0,64
    const int bm   = blockIdx.x * 128;
    const int bn   = blockIdx.y * 128;

    v8f acc[2][4];
#pragma unroll
    for (int mi = 0; mi < 2; ++mi)
#pragma unroll
        for (int g = 0; g < 4; ++g) acc[mi][g] = (v8f)0.0f;

    for (int k0 = 0; k0 < K; k0 += 32) {
        // Stage A and W tiles: 128 rows x 32 cols bf16 each = 512 uint4 each;
        // 2 async b128 copies per thread per tile, no VGPR data staging.
#pragma unroll
        for (int j = 0; j < 2; ++j) {
            int i = tid + 256 * j;
            int r = i >> 2, c = (i & 3) * 8;
            async_b128(A + (size_t)(bm + r) * K + k0 + c, As + r * 32 + c);
            async_b128(W + (size_t)(bn + r) * K + k0 + c, Ws + r * 32 + c);
        }
        wait_async0();
        __syncthreads();

        v16bf af[2];
#pragma unroll
        for (int mi = 0; mi < 2; ++mi)
            af[mi] = load_frag16(As, wm + mi * 16, 32);
#pragma unroll
        for (int g = 0; g < 4; ++g) {
            v16bf b    = load_frag16(Ws, wn + g * 16, 32);
            acc[0][g]  = wmma_bf16(af[0], b, acc[0][g]);
            acc[1][g]  = wmma_bf16(af[1], b, acc[1][g]);
        }
        __syncthreads();
    }

    // Epilogue. C layout: element r of v8f -> row = r + 8*(lane>=16), col = lane&15.
    const int hi = (lane >> 4) & 1;
    const int c0 = lane & 15;
#pragma unroll
    for (int mi = 0; mi < 2; ++mi) {
#pragma unroll
        for (int g = 0; g < 4; ++g) {
            int   col = bn + wn + g * 16 + c0;
            float bv  = bias[col];
#pragma unroll
            for (int r = 0; r < 8; ++r) {
                int   row = bm + wm + mi * 16 + r + 8 * hi;
                float v   = acc[mi][g][r] + bv;
                if (RELU) v = fmaxf(v, 0.0f);
                if (OUT_BF16)
                    ((unsigned short*)out)[(size_t)row * N + col] = f2bf(v);
                else
                    ((float*)out)[(size_t)row * N + col] = v;
            }
        }
    }
}

// ---------------------------------------------------------------------------
// Fused flash attention, one wave per (batch, head, 16-row q tile). dh = 64.
// ---------------------------------------------------------------------------
__global__ __launch_bounds__(32) void Transformer_34961033790145_attn(
    const unsigned short* __restrict__ Q, const unsigned short* __restrict__ Km,
    const unsigned short* __restrict__ V, const float* __restrict__ mask,
    unsigned short* __restrict__ O, int T, int Tk, int D, int causal,
    float inv_scale) {
    const int DH = 64;
    __shared__ unsigned short Ks[32 * 64];  // [key][dh] (async-staged)
    __shared__ unsigned short Vs[64 * 34];  // transposed: [dh][key], padded
    __shared__ unsigned short Ps[16 * 32];  // P tile, A-layout source

    const int lane = threadIdx.x;
    const int q0   = blockIdx.x * 16;
    const int head = blockIdx.y;
    const int b    = blockIdx.z;
    const int hi   = (lane >> 4) & 1;
    const int c0   = lane & 15;

    // Q fragments (16 x 64 -> two 16x32 A-frags), loaded once from global.
    const unsigned short* qbase = Q + (size_t)b * T * D + (size_t)head * DH;
    v16bf qf0 = load_frag16(qbase, q0, D);
    v16bf qf1 = load_frag16(qbase + 32, q0, D);

    float m[8], l[8];
    v8f   o[4];
#pragma unroll
    for (int r = 0; r < 8; ++r) { m[r] = -1e30f; l[r] = 0.0f; }
#pragma unroll
    for (int g = 0; g < 4; ++g) o[g] = (v8f)0.0f;

    const float INFP    = 1e10f;
    const int   nchunks = (Tk + 31) / 32;
    for (int kc = 0; kc < nchunks; ++kc) {
        const int kb = kc * 32;
        if (causal && kb > q0 + 15) break;

        // Stage K chunk (32 x 64) async; V chunk transposed via VGPRs.
#pragma unroll
        for (int j = 0; j < 8; ++j) {
            int idx = j * 32 + lane;
            int r = idx >> 3, c = (idx & 7) * 8;
            async_b128(Km + ((size_t)b * Tk + kb + r) * D + head * DH + c,
                       Ks + r * 64 + c);
            uint4 t =
                *(const uint4*)(V + ((size_t)b * Tk + kb + r) * D + head * DH + c);
            unsigned short tv[8];
            *(uint4*)tv = t;
#pragma unroll
            for (int e = 0; e < 8; ++e) Vs[(c + e) * 34 + r] = tv[e];
        }
        wait_async0();
        __syncthreads();

        // S = Q @ K^T for two 16-key tiles; chain over dh=64 (2 WMMAs each).
        v8f s0 = (v8f)0.0f, s1 = (v8f)0.0f;
        s0 = wmma_bf16(qf0, load_frag16(Ks, 0, 64), s0);
        s0 = wmma_bf16(qf1, load_frag16(Ks + 32, 0, 64), s0);
        s1 = wmma_bf16(qf0, load_frag16(Ks, 16, 64), s1);
        s1 = wmma_bf16(qf1, load_frag16(Ks + 32, 16, 64), s1);

        const int j0 = kb + c0;
        const int j1 = kb + 16 + c0;
        const float pen0 = (1.0f - mask[(size_t)b * Tk + j0]) * INFP;
        const float pen1 = (1.0f - mask[(size_t)b * Tk + j1]) * INFP;

#pragma unroll
        for (int r = 0; r < 8; ++r) {
            const int qrow = q0 + r + 8 * hi;
            float v0 = s0[r] - pen0 - ((causal && j0 > qrow) ? INFP : 0.0f);
            float v1 = s1[r] - pen1 - ((causal && j1 > qrow) ? INFP : 0.0f);
            v0 *= inv_scale;
            v1 *= inv_scale;

            // Row max across the 16 lanes of this half (ds_swizzle XOR tree).
            float mx = fmaxf(v0, v1);
            mx = fmaxf(mx, swzx<8>(mx));
            mx = fmaxf(mx, swzx<4>(mx));
            mx = fmaxf(mx, swzx<2>(mx));
            mx = fmaxf(mx, swzx<1>(mx));

            const float mn    = fmaxf(m[r], mx);
            const float alpha = __expf(m[r] - mn);
            m[r]              = mn;

            const float p0 = __expf(v0 - mn);
            const float p1 = __expf(v1 - mn);
            float sum = p0 + p1;
            sum += swzx<8>(sum);
            sum += swzx<4>(sum);
            sum += swzx<2>(sum);
            sum += swzx<1>(sum);
            l[r] = l[r] * alpha + sum;
#pragma unroll
            for (int g = 0; g < 4; ++g) o[g][r] *= alpha;

            Ps[(r + 8 * hi) * 32 + c0]      = f2bf(p0);
            Ps[(r + 8 * hi) * 32 + 16 + c0] = f2bf(p1);
        }
        __syncthreads();

        // O += P(16x32) @ V(32x64): 4 col-group WMMAs.
        v16bf pa = load_frag16(Ps, 0, 32);
#pragma unroll
        for (int g = 0; g < 4; ++g)
            o[g] = wmma_bf16(pa, load_frag16(Vs, g * 16, 34), o[g]);
        __syncthreads();
    }

    // Normalize and write bf16 output (feeds the output-projection GEMM).
#pragma unroll
    for (int r = 0; r < 8; ++r) {
        const float inv = 1.0f / l[r];
        const int qrow  = q0 + r + 8 * hi;
#pragma unroll
        for (int g = 0; g < 4; ++g) {
            O[((size_t)b * T + qrow) * D + head * DH + g * 16 + c0] =
                f2bf(o[g][r] * inv);
        }
    }
}

// ---------------------------------------------------------------------------
// h = LayerNorm(x + y); unbiased variance (ddof=1); eps added to std.
// Writes f32 (residual chain) and optional bf16 (next GEMM input).
// ---------------------------------------------------------------------------
__global__ __launch_bounds__(256) void Transformer_34961033790145_addln(
    const float* __restrict__ X, const float* __restrict__ Y,
    const float* __restrict__ gamma, const float* __restrict__ beta,
    float* __restrict__ out32, unsigned short* __restrict__ out16, int Dv) {
    const int    row = blockIdx.x;
    const float* x   = X + (size_t)row * Dv;
    const float* y   = Y + (size_t)row * Dv;
    const int tid = threadIdx.x, lane = tid & 31, w = tid >> 5;

    float s = 0.0f, s2 = 0.0f;
    for (int c = tid; c < Dv; c += 256) {
        float v = x[c] + y[c];
        s += v;
        s2 += v * v;
    }
    s += swzx<16>(s);  s2 += swzx<16>(s2);
    s += swzx<8>(s);   s2 += swzx<8>(s2);
    s += swzx<4>(s);   s2 += swzx<4>(s2);
    s += swzx<2>(s);   s2 += swzx<2>(s2);
    s += swzx<1>(s);   s2 += swzx<1>(s2);

    __shared__ float rs[8], rs2[8];
    if (!lane) { rs[w] = s; rs2[w] = s2; }
    __syncthreads();
    float ts = 0.0f, ts2 = 0.0f;
#pragma unroll
    for (int i = 0; i < 8; ++i) { ts += rs[i]; ts2 += rs2[i]; }

    const float mean  = ts / Dv;
    const float var   = (ts2 - mean * ts) / (Dv - 1);
    const float denom = sqrtf(fmaxf(var, 0.0f)) + 1e-6f;

    for (int c = tid; c < Dv; c += 256) {
        float v = x[c] + y[c];
        float r = gamma[c] * (v - mean) / denom + beta[c];
        out32[(size_t)row * Dv + c] = r;
        if (out16) out16[(size_t)row * Dv + c] = f2bf(r);
    }
}

// ---------------------------------------------------------------------------
// Host orchestration
// ---------------------------------------------------------------------------
extern "C" void kernel_launch(void* const* d_in, const int* in_sizes, int n_in,
                              void* d_out, int out_size, void* d_ws,
                              size_t ws_size, hipStream_t stream) {
    (void)in_sizes; (void)n_in; (void)out_size;
    const int B = 4, T = 1024, D = 1024, NH = 16, H = 4096;
    const int M = B * T;

    const float* x        = (const float*)d_in[0];
    const float* enc      = (const float*)d_in[1];
    const float* mask_src = (const float*)d_in[2];
    const float* mask_trg = (const float*)d_in[3];
    const float* sa_w[4]  = {(const float*)d_in[4], (const float*)d_in[6],
                             (const float*)d_in[8], (const float*)d_in[10]};
    const float* sa_b[4]  = {(const float*)d_in[5], (const float*)d_in[7],
                             (const float*)d_in[9], (const float*)d_in[11]};
    const float* sa_gamma = (const float*)d_in[12];
    const float* sa_beta  = (const float*)d_in[13];
    const float* ca_w[4]  = {(const float*)d_in[14], (const float*)d_in[16],
                             (const float*)d_in[18], (const float*)d_in[20]};
    const float* ca_b[4]  = {(const float*)d_in[15], (const float*)d_in[17],
                             (const float*)d_in[19], (const float*)d_in[21]};
    const float* ca_gamma = (const float*)d_in[22];
    const float* ca_beta  = (const float*)d_in[23];
    const float* ff_w1 = (const float*)d_in[24];
    const float* ff_b1 = (const float*)d_in[25];
    const float* ff_w2 = (const float*)d_in[26];
    const float* ff_b2 = (const float*)d_in[27];
    const float* ff_gamma = (const float*)d_in[28];
    const float* ff_beta  = (const float*)d_in[29];

    // Workspace carve-up.
    const size_t DD = (size_t)D * D, DH = (size_t)D * H;
    const size_t MD = (size_t)M * D, MH = (size_t)M * H;
    size_t off = 0;
    auto take = [&](size_t bytes) -> size_t {
        size_t o = off;
        off += (bytes + 255) & ~(size_t)255;
        return o;
    };
    size_t o_wsa[4], o_wca[4];
    for (int i = 0; i < 4; ++i) o_wsa[i] = take(DD * 2);
    for (int i = 0; i < 4; ++i) o_wca[i] = take(DD * 2);
    size_t o_wff1 = take(DH * 2), o_wff2 = take(DH * 2);
    size_t o_xb = take(MD * 2), o_encb = take(MD * 2);
    size_t o_qb = take(MD * 2), o_kb = take(MD * 2), o_vb = take(MD * 2);
    size_t o_ob = take(MD * 2), o_h1b = take(MD * 2), o_h2b = take(MD * 2);
    size_t o_ffb = take(MH * 2);
    size_t o_h1 = take(MD * 4), o_h2 = take(MD * 4), o_tmp = take(MD * 4);
    if (off > ws_size) return;  // insufficient scratch: bail deterministically

    char* ws = (char*)d_ws;
    auto U16 = [&](size_t o) { return (unsigned short*)(ws + o); };
    auto F32 = [&](size_t o) { return (float*)(ws + o); };

    const dim3 cblk(256), cgrd(1024);
    auto cvt = [&](const float* s, unsigned short* dptr, size_t n) {
        Transformer_34961033790145_cvt<<<cgrd, cblk, 0, stream>>>(s, dptr, (int)n);
    };
    cvt(x, U16(o_xb), MD);
    cvt(enc, U16(o_encb), MD);
    for (int i = 0; i < 4; ++i) cvt(sa_w[i], U16(o_wsa[i]), DD);
    for (int i = 0; i < 4; ++i) cvt(ca_w[i], U16(o_wca[i]), DD);
    cvt(ff_w1, U16(o_wff1), DH);
    cvt(ff_w2, U16(o_wff2), DH);

    const dim3 gDD(M / 128, D / 128), gDH(M / 128, H / 128), gblk(256);
    const dim3 agrd(T / 16, NH, B), ablk(32);
    const float inv_scale = 1.0f / 32.0f;  // 1/sqrt(D), D=1024

    // ---- Causal self-attention block ----
    Transformer_34961033790145_gemm<1, 0><<<gDD, gblk, 0, stream>>>(
        U16(o_xb), U16(o_wsa[0]), sa_b[0], U16(o_qb), M, D, D);
    Transformer_34961033790145_gemm<1, 0><<<gDD, gblk, 0, stream>>>(
        U16(o_xb), U16(o_wsa[1]), sa_b[1], U16(o_kb), M, D, D);
    Transformer_34961033790145_gemm<1, 0><<<gDD, gblk, 0, stream>>>(
        U16(o_xb), U16(o_wsa[2]), sa_b[2], U16(o_vb), M, D, D);
    Transformer_34961033790145_attn<<<agrd, ablk, 0, stream>>>(
        U16(o_qb), U16(o_kb), U16(o_vb), mask_trg, U16(o_ob), T, T, D, 1,
        inv_scale);
    Transformer_34961033790145_gemm<0, 0><<<gDD, gblk, 0, stream>>>(
        U16(o_ob), U16(o_wsa[3]), sa_b[3], F32(o_tmp), M, D, D);
    Transformer_34961033790145_addln<<<M, gblk, 0, stream>>>(
        x, F32(o_tmp), sa_gamma, sa_beta, F32(o_h1), U16(o_h1b), D);

    // ---- Cross-attention block ----
    Transformer_34961033790145_gemm<1, 0><<<gDD, gblk, 0, stream>>>(
        U16(o_h1b), U16(o_wca[0]), ca_b[0], U16(o_qb), M, D, D);
    Transformer_34961033790145_gemm<1, 0><<<gDD, gblk, 0, stream>>>(
        U16(o_encb), U16(o_wca[1]), ca_b[1], U16(o_kb), M, D, D);
    Transformer_34961033790145_gemm<1, 0><<<gDD, gblk, 0, stream>>>(
        U16(o_encb), U16(o_wca[2]), ca_b[2], U16(o_vb), M, D, D);
    Transformer_34961033790145_attn<<<agrd, ablk, 0, stream>>>(
        U16(o_qb), U16(o_kb), U16(o_vb), mask_src, U16(o_ob), T, T, D, 0,
        inv_scale);
    Transformer_34961033790145_gemm<0, 0><<<gDD, gblk, 0, stream>>>(
        U16(o_ob), U16(o_wca[3]), ca_b[3], F32(o_tmp), M, D, D);
    Transformer_34961033790145_addln<<<M, gblk, 0, stream>>>(
        F32(o_h1), F32(o_tmp), ca_gamma, ca_beta, F32(o_h2), U16(o_h2b), D);

    // ---- Feed-forward block ----
    Transformer_34961033790145_gemm<1, 1><<<gDH, gblk, 0, stream>>>(
        U16(o_h2b), U16(o_wff1), ff_b1, U16(o_ffb), M, H, D);
    Transformer_34961033790145_gemm<0, 0><<<gDD, gblk, 0, stream>>>(
        U16(o_ffb), U16(o_wff2), ff_b2, F32(o_tmp), M, D, H);
    Transformer_34961033790145_addln<<<M, gblk, 0, stream>>>(
        F32(o_h2), F32(o_tmp), ff_gamma, ff_beta, (float*)d_out, nullptr, D);
}
